// Block_28424093565016
// MI455X (gfx1250) — compile-verified, matrix operand order checked by reading, not run on hardware
//
#include <hip/hip_runtime.h>
#include <math.h>

// ---------------- problem constants ----------------
#define HDIM 768
#define NHEADS 12
#define HD 64
#define MLPD 3072
#define BATCH 16
#define SEQ 577
#define MROWS (BATCH * SEQ)      // 9232
#define NPAD 640                 // padded seq rows for q/v head buffers
#define BH (BATCH * NHEADS)      // 192
#define KVTILES 19               // ceil(577/32) -> covers cols 0..607
#define SCOLS 608                // KVTILES*32
#define SSTR 616                 // f32 LDS stride for score rows (pad, mult of 8)

typedef __attribute__((ext_vector_type(16))) __bf16 bf16x16;
typedef __attribute__((ext_vector_type(8)))  float  f32x8;

union Frag { bf16x16 v; uint4 q[2]; };

__device__ __forceinline__ f32x8 fzero() {
    f32x8 z = {0.f, 0.f, 0.f, 0.f, 0.f, 0.f, 0.f, 0.f};
    return z;
}

__device__ __forceinline__ f32x8 wmma_bf16(const Frag& a, const Frag& b, f32x8 c) {
    // D = A(16x32 bf16) x B(32x16 bf16) + C(f32)
    return __builtin_amdgcn_wmma_f32_16x16x32_bf16(
        false, a.v, false, b.v, (short)0, c, false, false);
}

// ---------------- fp32 -> bf16 transpose-convert (one-time per weight) ----------
// src[K][N] f32 -> dst[N][K] bf16, so GEMM B-tile staging is fully vectorized.
__global__ __launch_bounds__(256)
void cvt_transpose_bf16(const float* __restrict__ src, __bf16* __restrict__ dst,
                        int K, int N) {
    __shared__ float tile[32][33];
    int bn = blockIdx.x * 32;            // n base
    int bk = blockIdx.y * 32;            // k base
    int tx = threadIdx.x & 31;
    int ty = threadIdx.x >> 5;           // 0..7
#pragma unroll
    for (int i = 0; i < 32; i += 8)
        tile[ty + i][tx] = src[(size_t)(bk + ty + i) * N + bn + tx];
    __syncthreads();
#pragma unroll
    for (int i = 0; i < 32; i += 8)
        dst[(size_t)(bn + ty + i) * K + bk + tx] = (__bf16)tile[tx][ty + i];
}

// ---------------- LayerNorm (768 cols) -> bf16 ----------------
__global__ __launch_bounds__(256)
void layernorm_bf16(const float* __restrict__ x, const float* __restrict__ g,
                    const float* __restrict__ b, __bf16* __restrict__ y) {
    __shared__ float red[256];
    int row = blockIdx.x;
    int t = threadIdx.x;
    const float* xr = x + (size_t)row * HDIM;
    float v0 = xr[t], v1 = xr[t + 256], v2 = xr[t + 512];
    red[t] = v0 + v1 + v2;
    __syncthreads();
    for (int o = 128; o > 0; o >>= 1) { if (t < o) red[t] += red[t + o]; __syncthreads(); }
    float mu = red[0] * (1.0f / HDIM);
    __syncthreads();
    float d0 = v0 - mu, d1 = v1 - mu, d2 = v2 - mu;
    red[t] = d0 * d0 + d1 * d1 + d2 * d2;
    __syncthreads();
    for (int o = 128; o > 0; o >>= 1) { if (t < o) red[t] += red[t + o]; __syncthreads(); }
    float rstd = rsqrtf(red[0] * (1.0f / HDIM) + 1e-6f);
    __bf16* yr = y + (size_t)row * HDIM;
    yr[t]       = (__bf16)(d0 * rstd * g[t]       + b[t]);
    yr[t + 256] = (__bf16)(d1 * rstd * g[t + 256] + b[t + 256]);
    yr[t + 512] = (__bf16)(d2 * rstd * g[t + 512] + b[t + 512]);
}

// ---------------- generic bf16 WMMA GEMM (B pre-transposed: Bt[N][K]) ----------
// Double-buffered LDS staging: tile k+1 global loads are issued before the tile-k
// WMMAs, stored to the alternate buffer after, one barrier per k-tile.
#define EPI_QV          0   // bf16 out, head layout; out2 (optional) = transposed head layout
#define EPI_BIAS_RESID  1   // f32 out row-major, + resid
#define EPI_GELU        2   // bf16 out row-major, gelu(acc+bias)

template <int EPI>
__global__ __launch_bounds__(256)
void gemm_bf16(const __bf16* __restrict__ A, const __bf16* __restrict__ Bt,
               const float* __restrict__ bias, const float* __restrict__ resid,
               void* __restrict__ out, void* __restrict__ out2,
               int Mdim, int Kdim, int Ndim, float scale) {
    constexpr int BM = 128, BN = 128, BK = 64, SA = 72;
    __shared__ __align__(16) __bf16 sA[2][BM * SA];   // A rows  [m][k], ping-pong
    __shared__ __align__(16) __bf16 sB[2][BN * SA];   // Bt rows [n][k], ping-pong

    int tid  = threadIdx.x;
    int lane = tid & 31, wave = tid >> 5;
    int wm = wave >> 1, wn = wave & 1;                // 4 x 2 wave grid
    int lr = lane & 15, lh = lane >> 4;
    int block_m = blockIdx.y * BM, block_n = blockIdx.x * BN;

    f32x8 acc[2][4];
#pragma unroll
    for (int i = 0; i < 2; ++i)
#pragma unroll
        for (int j = 0; j < 4; ++j) acc[i][j] = fzero();

    int r = tid >> 1;                 // 0..127
    int c = (tid & 1) << 5;           // 0 or 32
    const __bf16* baseA = A  + (size_t)(block_m + r) * Kdim + c;
    const __bf16* baseB = Bt + (size_t)(block_n + r) * Kdim + c;

    // prologue: stage tile 0 into buffer 0
    {
        uint4 a0 = *(const uint4*)(baseA);
        uint4 a1 = *(const uint4*)(baseA + 8);
        uint4 a2 = *(const uint4*)(baseA + 16);
        uint4 a3 = *(const uint4*)(baseA + 24);
        uint4 b0 = *(const uint4*)(baseB);
        uint4 b1 = *(const uint4*)(baseB + 8);
        uint4 b2 = *(const uint4*)(baseB + 16);
        uint4 b3 = *(const uint4*)(baseB + 24);
        *(uint4*)(&sA[0][r * SA + c])      = a0;
        *(uint4*)(&sA[0][r * SA + c + 8])  = a1;
        *(uint4*)(&sA[0][r * SA + c + 16]) = a2;
        *(uint4*)(&sA[0][r * SA + c + 24]) = a3;
        *(uint4*)(&sB[0][r * SA + c])      = b0;
        *(uint4*)(&sB[0][r * SA + c + 8])  = b1;
        *(uint4*)(&sB[0][r * SA + c + 16]) = b2;
        *(uint4*)(&sB[0][r * SA + c + 24]) = b3;
    }
    __syncthreads();

    int nk = Kdim / BK;               // Kdim is a multiple of 64
    for (int kt = 0; kt < nk; ++kt) {
        int cur = kt & 1, nxt = cur ^ 1;
        bool more = (kt + 1 < nk);

        // issue next tile's global loads (latency hidden behind 16 WMMAs)
        uint4 a0, a1, a2, a3, b0, b1, b2, b3;
        if (more) {
            const __bf16* srcA = baseA + (size_t)(kt + 1) * BK;
            const __bf16* srcB = baseB + (size_t)(kt + 1) * BK;
            a0 = *(const uint4*)(srcA);
            a1 = *(const uint4*)(srcA + 8);
            a2 = *(const uint4*)(srcA + 16);
            a3 = *(const uint4*)(srcA + 24);
            b0 = *(const uint4*)(srcB);
            b1 = *(const uint4*)(srcB + 8);
            b2 = *(const uint4*)(srcB + 16);
            b3 = *(const uint4*)(srcB + 24);
            if (kt + 2 < nk) {        // pull tile k+2 toward L2/L0
                __builtin_prefetch(srcA + BK, 0, 0);
                __builtin_prefetch(srcB + BK, 0, 0);
            }
        }

        // WMMAs on current buffer
#pragma unroll
        for (int ks = 0; ks < 2; ++ks) {
            Frag a[2], b[4];
#pragma unroll
            for (int i = 0; i < 2; ++i) {
                const __bf16* p = &sA[cur][(wm * 32 + i * 16 + lr) * SA + ks * 32 + lh * 8];
                a[i].q[0] = *(const uint4*)(p);
                a[i].q[1] = *(const uint4*)(p + 16);
            }
#pragma unroll
            for (int j = 0; j < 4; ++j) {
                const __bf16* p = &sB[cur][(wn * 64 + j * 16 + lr) * SA + ks * 32 + lh * 8];
                b[j].q[0] = *(const uint4*)(p);
                b[j].q[1] = *(const uint4*)(p + 16);
            }
#pragma unroll
            for (int i = 0; i < 2; ++i)
#pragma unroll
                for (int j = 0; j < 4; ++j)
                    acc[i][j] = wmma_bf16(a[i], b[j], acc[i][j]);
        }

        // commit next tile into the alternate buffer
        if (more) {
            *(uint4*)(&sA[nxt][r * SA + c])      = a0;
            *(uint4*)(&sA[nxt][r * SA + c + 8])  = a1;
            *(uint4*)(&sA[nxt][r * SA + c + 16]) = a2;
            *(uint4*)(&sA[nxt][r * SA + c + 24]) = a3;
            *(uint4*)(&sB[nxt][r * SA + c])      = b0;
            *(uint4*)(&sB[nxt][r * SA + c + 8])  = b1;
            *(uint4*)(&sB[nxt][r * SA + c + 16]) = b2;
            *(uint4*)(&sB[nxt][r * SA + c + 24]) = b3;
        }
        __syncthreads();
    }

    // epilogue
#pragma unroll
    for (int i = 0; i < 2; ++i) {
        int rbase = block_m + wm * 32 + i * 16 + lh * 8;
#pragma unroll
        for (int j = 0; j < 4; ++j) {
            int col = block_n + wn * 64 + j * 16 + lr;
            float bb = bias[col];
#pragma unroll
            for (int g = 0; g < 8; ++g) {
                int row = rbase + g;
                if (row < Mdim) {
                    float vacc = acc[i][j][g] + bb;
                    if constexpr (EPI == EPI_QV) {
                        int bidx = row / SEQ;
                        int nidx = row - bidx * SEQ;
                        int h = col >> 6, d = col & 63;
                        __bf16 val = (__bf16)(vacc * scale);
                        ((__bf16*)out)[((size_t)(bidx * NHEADS + h) * NPAD + nidx) * HD + d] = val;
                        if (out2)   // transposed copy for barrier-free attention pass C
                            ((__bf16*)out2)[((size_t)(bidx * NHEADS + h) * HD + d) * NPAD + nidx] = val;
                    } else if constexpr (EPI == EPI_BIAS_RESID) {
                        ((float*)out)[(size_t)row * Ndim + col] =
                            vacc + resid[(size_t)row * Ndim + col];
                    } else {  // EPI_GELU (exact)
                        float gx = 0.5f * vacc * (1.0f + erff(vacc * 0.70710678f));
                        ((__bf16*)out)[(size_t)row * Ndim + col] = (__bf16)gx;
                    }
                }
            }
        }
    }
}

// ---------------- attention: S = Q Vt (Q prescaled), softmax, O = P V -----------
// One workgroup (4 waves) per (b*h, 64-query tile). Full 64x608 f32 score tile in
// LDS (~158KB, CDNA5 320KB/WGP), softmax in place (bf16 P aliases S), then
// barrier-free WMMA P x V with B-fragments streamed from the transposed V copy.
__global__ __launch_bounds__(128)
void attn_kernel(const __bf16* __restrict__ qb, const __bf16* __restrict__ vb,
                 const __bf16* __restrict__ vtb, __bf16* __restrict__ ob) {
    extern __shared__ char smem[];
    float*  S = (float*)smem;                        // [64][SSTR] f32
    __bf16* P = (__bf16*)smem;                       // aliases S, row stride 2*SSTR

    int bh = blockIdx.y;
    int qt = blockIdx.x;
    int tid = threadIdx.x, lane = tid & 31, wave = tid >> 5;
    int lr = lane & 15, lh = lane >> 4;
    const __bf16* qh  = qb  + (size_t)bh * NPAD * HD;
    const __bf16* vh  = vb  + (size_t)bh * NPAD * HD;
    const __bf16* vth = vtb + (size_t)bh * HD * NPAD;

    // ---- pass A: scores (A=Q rows, B=V^T columns == V rows, direct from global) --
    Frag aq[2];
    int qrow = qt * 64 + wave * 16 + lr;             // < NPAD=640 always
#pragma unroll
    for (int ks = 0; ks < 2; ++ks) {
        const __bf16* p = qh + (size_t)qrow * HD + ks * 32 + lh * 8;
        aq[ks].q[0] = *(const uint4*)(p);
        aq[ks].q[1] = *(const uint4*)(p + 16);
    }
    for (int j = 0; j < KVTILES; ++j) {
        int kv0 = j * 32;
        f32x8 s0 = fzero(), s1 = fzero();
#pragma unroll
        for (int ks = 0; ks < 2; ++ks) {
            Frag b0, b1;
            const __bf16* p0 = vh + (size_t)(kv0 + lr) * HD + ks * 32 + lh * 8;
            const __bf16* p1 = p0 + 16 * HD;
            b0.q[0] = *(const uint4*)(p0); b0.q[1] = *(const uint4*)(p0 + 16);
            b1.q[0] = *(const uint4*)(p1); b1.q[1] = *(const uint4*)(p1 + 16);
            s0 = wmma_bf16(aq[ks], b0, s0);
            s1 = wmma_bf16(aq[ks], b1, s1);
        }
        int srow = wave * 16 + lh * 8;
        int c0 = kv0 + lr, c1 = kv0 + 16 + lr;
        bool in0 = (c0 < SEQ), in1 = (c1 < SEQ);
#pragma unroll
        for (int g = 0; g < 8; ++g) {
            S[(srow + g) * SSTR + c0] = in0 ? s0[g] : -1e30f;
            S[(srow + g) * SSTR + c1] = in1 ? s1[g] : -1e30f;
        }
    }
    __syncthreads();

    // ---- pass B: softmax, in place (bf16 P overwrites f32 S, same thread/row) ----
    if (tid < 64) {
        float* sr = S + (size_t)tid * SSTR;
        float mx = -1e30f;
        for (int cc = 0; cc < SCOLS; ++cc) mx = fmaxf(mx, sr[cc]);
        float sum = 0.0f;
        for (int cc = 0; cc < SCOLS; ++cc) { float e = __expf(sr[cc] - mx); sr[cc] = e; sum += e; }
        float rs = 1.0f / sum;
        __bf16* pr = P + (size_t)tid * (2 * SSTR);
        for (int cc = 0; cc < SCOLS; ++cc) pr[cc] = (__bf16)(sr[cc] * rs);
    }
    __syncthreads();

    // ---- pass C: O = P x V (B-frags straight from transposed V; no barriers) ----
    f32x8 o[4];
#pragma unroll
    for (int f = 0; f < 4; ++f) o[f] = fzero();

    for (int j = 0; j < KVTILES; ++j) {
        int kv0 = j * 32;
        Frag ap;
        const __bf16* pp = P + (size_t)(wave * 16 + lr) * (2 * SSTR) + kv0 + lh * 8;
        ap.q[0] = *(const uint4*)(pp);
        ap.q[1] = *(const uint4*)(pp + 16);
#pragma unroll
        for (int f = 0; f < 4; ++f) {
            Frag bv;
            const __bf16* vp = vth + (size_t)(f * 16 + lr) * NPAD + kv0 + lh * 8;
            bv.q[0] = *(const uint4*)(vp);
            bv.q[1] = *(const uint4*)(vp + 16);
            o[f] = wmma_bf16(ap, bv, o[f]);
        }
    }

    // store O -> row-major [M x 768] bf16 (ready for Wo GEMM)
    int b = bh / NHEADS, h = bh % NHEADS;
    int orow0 = qt * 64 + wave * 16 + lh * 8;
#pragma unroll
    for (int f = 0; f < 4; ++f) {
        int d = f * 16 + lr;
#pragma unroll
        for (int g = 0; g < 8; ++g) {
            int n = orow0 + g;
            if (n < SEQ)
                ob[(size_t)(b * SEQ + n) * HDIM + h * HD + d] = (__bf16)o[f][g];
        }
    }
}

// ---------------- host launcher ----------------
extern "C" void kernel_launch(void* const* d_in, const int* in_sizes, int n_in,
                              void* d_out, int out_size, void* d_ws, size_t ws_size,
                              hipStream_t stream) {
    const float* x    = (const float*)d_in[0];
    const float* Wq   = (const float*)d_in[1];
    const float* bq   = (const float*)d_in[2];
    // d_in[3], d_in[4] = Wk, bk : unused (reference computes Q@V^T, K is dead)
    const float* Wv   = (const float*)d_in[5];
    const float* bv   = (const float*)d_in[6];
    const float* Wo   = (const float*)d_in[7];
    const float* bo   = (const float*)d_in[8];
    const float* W1   = (const float*)d_in[9];
    const float* b1   = (const float*)d_in[10];
    const float* W2   = (const float*)d_in[11];
    const float* b2   = (const float*)d_in[12];
    const float* ln1g = (const float*)d_in[13];
    const float* ln1b = (const float*)d_in[14];
    const float* ln2g = (const float*)d_in[15];
    const float* ln2b = (const float*)d_in[16];
    (void)in_sizes; (void)n_in; (void)out_size; (void)ws_size;

    // workspace carve (all intermediates live in the 192MB-L2-sized ws)
    char* ws = (char*)d_ws;
    size_t off = 0;
    auto carve = [&](size_t bytes) {
        char* p = ws + off;
        off = (off + bytes + 255) & ~(size_t)255;
        return p;
    };
    __bf16* y_bf   = (__bf16*)carve((size_t)MROWS * HDIM * 2);   // reused as y2
    __bf16* q_buf  = (__bf16*)carve((size_t)BH * NPAD * HD * 2);
    __bf16* v_buf  = (__bf16*)carve((size_t)BH * NPAD * HD * 2);
    __bf16* vT_buf = (__bf16*)carve((size_t)BH * HD * NPAD * 2);
    __bf16* o_bf   = (__bf16*)carve((size_t)MROWS * HDIM * 2);
    float*  x1     = (float*) carve((size_t)MROWS * HDIM * 4);
    __bf16* m_bf   = (__bf16*)carve((size_t)MROWS * MLPD * 2);
    __bf16* wqT    = (__bf16*)carve((size_t)HDIM * HDIM * 2);    // [N][K] bf16
    __bf16* wvT    = (__bf16*)carve((size_t)HDIM * HDIM * 2);
    __bf16* woT    = (__bf16*)carve((size_t)HDIM * HDIM * 2);
    __bf16* w1T    = (__bf16*)carve((size_t)MLPD * HDIM * 2);    // [3072][768]
    __bf16* w2T    = (__bf16*)carve((size_t)HDIM * MLPD * 2);    // [768][3072]
    (void)carve((size_t)1 << 20);   // slack for benign OOB tile reads on M tail

    // 1) weights -> bf16, transposed once (amortized over all M tiles)
    dim3 thh(HDIM / 32, HDIM / 32);      // src K x N = 768 x 768
    cvt_transpose_bf16<<<thh, 256, 0, stream>>>(Wq, wqT, HDIM, HDIM);
    cvt_transpose_bf16<<<thh, 256, 0, stream>>>(Wv, wvT, HDIM, HDIM);
    cvt_transpose_bf16<<<thh, 256, 0, stream>>>(Wo, woT, HDIM, HDIM);
    dim3 t1(MLPD / 32, HDIM / 32);       // W1: K=768, N=3072
    cvt_transpose_bf16<<<t1, 256, 0, stream>>>(W1, w1T, HDIM, MLPD);
    dim3 t2(HDIM / 32, MLPD / 32);       // W2: K=3072, N=768
    cvt_transpose_bf16<<<t2, 256, 0, stream>>>(W2, w2T, MLPD, HDIM);

    // zero q/v pads so padded attention rows are exact zeros
    hipMemsetAsync(q_buf,  0, (size_t)BH * NPAD * HD * 2, stream);
    hipMemsetAsync(v_buf,  0, (size_t)BH * NPAD * HD * 2, stream);
    hipMemsetAsync(vT_buf, 0, (size_t)BH * HD * NPAD * 2, stream);

    // 2) LN1
    layernorm_bf16<<<MROWS, 256, 0, stream>>>(x, ln1g, ln1b, y_bf);

    // 3) Q (scaled by 1/sqrt(64)) and V projections -> head layout (+V^T copy)
    dim3 g768(HDIM / 128, (MROWS + 127) / 128);
    gemm_bf16<EPI_QV><<<g768, 256, 0, stream>>>(y_bf, wqT, bq, nullptr, q_buf, nullptr,
                                                MROWS, HDIM, HDIM, 0.125f);
    gemm_bf16<EPI_QV><<<g768, 256, 0, stream>>>(y_bf, wvT, bv, nullptr, v_buf, vT_buf,
                                                MROWS, HDIM, HDIM, 1.0f);

    // 4) attention (scores = Q@V^T per reference)
    dim3 ga((SEQ + 63) / 64, BH);
    size_t attn_lds = (size_t)64 * SSTR * 4;   // ~158 KB
    attn_kernel<<<ga, 128, attn_lds, stream>>>(q_buf, v_buf, vT_buf, o_bf);

    // 5) output projection + residual -> x1 (f32)
    gemm_bf16<EPI_BIAS_RESID><<<g768, 256, 0, stream>>>(o_bf, woT, bo, x, x1, nullptr,
                                                        MROWS, HDIM, HDIM, 1.0f);

    // 6) LN2 (reuse y_bf)
    layernorm_bf16<<<MROWS, 256, 0, stream>>>(x1, ln2g, ln2b, y_bf);

    // 7) MLP
    dim3 g3072(MLPD / 128, (MROWS + 127) / 128);
    gemm_bf16<EPI_GELU><<<g3072, 256, 0, stream>>>(y_bf, w1T, b1, nullptr, m_bf, nullptr,
                                                   MROWS, HDIM, MLPD, 1.0f);
    gemm_bf16<EPI_BIAS_RESID><<<g768, 256, 0, stream>>>(m_bf, w2T, b2, x1, d_out, nullptr,
                                                        MROWS, MLPD, HDIM, 1.0f);
}
